// RNN_16020228014568
// MI455X (gfx1250) — matrix operand order repaired and checked by reading, not compiled, not run on hardware
//
#include <hip/hip_runtime.h>
#include <hip/hip_bf16.h>

// RNN forward for MI455X (gfx1250, wave32, WMMA).
// Phase 1: xw[t,b,h] = x[t,b,:].w_ih[h,:] + b_ih[h] + b_hh[h]  -> written into d_out
// Phase 2: for t: h = tanh(xw[t] + h @ w_hh^T), overwriting d_out[t] with h[t].
//          Batch rows are independent -> 4 workgroups of M=16, each sequential in t,
//          with w_hh (f16) and h (f16) resident in LDS; w_hh B-fragments live in VGPRs.

#define SEQ   2048
#define BATCH 64
#define DIN   256
#define HDIM  256
#define LDK   264   // padded LDS row stride (halves): 16B skew/row -> conflict-free b128 frag loads

typedef __attribute__((ext_vector_type(4)))  _Float16 v4h;
typedef __attribute__((ext_vector_type(8)))  _Float16 v8h;
typedef __attribute__((ext_vector_type(16))) _Float16 v16h;
typedef __attribute__((ext_vector_type(8)))  float    v8f;

// A fragment (16xK=32, f16). ISA layout: lanes 0-15 hold row M=lane with K {0..7,16..23},
// lanes 16-31 hold row M=lane-16 with K {8..15,24..31}. Two 16B LDS reads per lane.
__device__ __forceinline__ v16h ldsA(const _Float16* base, int row0, int k0, int lane) {
  const int r  = lane & 15;
  const int kh = (lane >> 4) << 3;              // 0 or 8
  const _Float16* p = base + (row0 + r) * LDK + k0 + kh;
  v8h lo = *(const v8h*)p;                      // K = k0+kh .. +7
  v8h hi = *(const v8h*)(p + 16);               // K = k0+kh+16 .. +23
  return __builtin_shufflevector(lo, hi, 0,1,2,3,4,5,6,7,8,9,10,11,12,13,14,15);
}

// B fragment (K=32 x 16, f16) with B stored row-major as [N][K] (i.e. W[h][k]).
// Lanes 0-15: col N=lane, K {0..7,16..23}; lanes 16-31: col N=lane-16, K {8..15,24..31}.
__device__ __forceinline__ v16h ldsB(const _Float16* base, int n0, int k0, int lane) {
  const int n  = lane & 15;
  const int kh = (lane >> 4) << 3;              // 0 or 8
  const _Float16* p = base + (n0 + n) * LDK + k0 + kh;
  v8h lo = *(const v8h*)p;
  v8h hi = *(const v8h*)(p + 16);
  return __builtin_shufflevector(lo, hi, 0,1,2,3,4,5,6,7,8,9,10,11,12,13,14,15);
}

__device__ __forceinline__ v8f wmma_f16(v16h a, v16h b, v8f c) {
  return __builtin_amdgcn_wmma_f32_16x16x32_f16(false, a, false, b, (short)0, c, false, false);
}

// ---------------- Phase 1: input projection GEMM ----------------
// Grid: (T*B)/64 blocks x 256 threads (8 waves). Block = 64 rows x all 256 cols.
// Wave w owns N-strip [32w, 32w+32): 4 M-tiles x 2 N-tiles = 8 accumulators, K-loop of 8.
// __launch_bounds__(256): 2 waves/SIMD -> full VGPR budget, no spills.
__global__ __launch_bounds__(256) void rnn_xproj(const float* __restrict__ x,
                                                 const float* __restrict__ w_ih,
                                                 const float* __restrict__ b_ih,
                                                 const float* __restrict__ b_hh,
                                                 float* __restrict__ out) {
  extern __shared__ __align__(16) char smem[];
  _Float16* sW = (_Float16*)smem;                       // [HDIM][LDK]  w_ih as [N][K], f16
  _Float16* sX = (_Float16*)(smem + HDIM * LDK * 2);    // [64][LDK]    x tile  [M][K], f16
  const int  tid  = threadIdx.x;
  const int  lane = tid & 31;
  const int  wave = tid >> 5;
  const long m0   = (long)blockIdx.x * 64;

  // Cooperative f32->f16 staging (vectorized 4-wide)
  for (int i = tid * 4; i < HDIM * DIN; i += 256 * 4) {
    float4 v = *(const float4*)(w_ih + i);
    v4h h = {(_Float16)v.x, (_Float16)v.y, (_Float16)v.z, (_Float16)v.w};
    *(v4h*)(sW + (i >> 8) * LDK + (i & 255)) = h;
  }
  for (int i = tid * 4; i < 64 * DIN; i += 256 * 4) {
    float4 v = *(const float4*)(x + m0 * DIN + i);
    v4h h = {(_Float16)v.x, (_Float16)v.y, (_Float16)v.z, (_Float16)v.w};
    *(v4h*)(sX + (i >> 8) * LDK + (i & 255)) = h;
  }
  __syncthreads();

  const int n0 = wave * 32;
  const int nl = lane & 15;
  const int rb = (lane >> 4) << 3;                      // C/D: VGPR r -> row rb+r
  const float bias0 = b_ih[n0 + nl]      + b_hh[n0 + nl];
  const float bias1 = b_ih[n0 + 16 + nl] + b_hh[n0 + 16 + nl];

  v8f acc[4][2] = {};
#pragma unroll
  for (int kb = 0; kb < 8; ++kb) {
    v16h bf0 = ldsB(sW, n0,      kb * 32, lane);
    v16h bf1 = ldsB(sW, n0 + 16, kb * 32, lane);
#pragma unroll
    for (int mt = 0; mt < 4; ++mt) {
      v16h af = ldsA(sX, mt * 16, kb * 32, lane);
      acc[mt][0] = wmma_f16(af, bf0, acc[mt][0]);
      acc[mt][1] = wmma_f16(af, bf1, acc[mt][1]);
    }
  }

#pragma unroll
  for (int mt = 0; mt < 4; ++mt) {
    const long rowb = m0 + mt * 16 + rb;
#pragma unroll
    for (int r = 0; r < 8; ++r) {
      out[(rowb + r) * HDIM + (n0 + nl)]      = acc[mt][0][r] + bias0;
      out[(rowb + r) * HDIM + (n0 + 16 + nl)] = acc[mt][1][r] + bias1;
    }
  }
}

// ---------------- Phase 2: sequential recurrence ----------------
// Grid: 4 blocks x 512 threads (16 waves). Block owns batch rows [16*bid, 16*bid+16).
// Wave w owns N-tile [16w, 16w+16); its 8 w_hh B-fragments are preloaded into VGPRs.
// Per step: 8-deep WMMA chain over K=256 from h (f16, LDS), + xw (independent global
// loads, hoisted by the scheduler to overlap the chain; L2-resident), tanh, write h to
// LDS (f16) for next step and to d_out (f32).
// __launch_bounds__(512): 4 waves/SIMD -> ~128 VGPRs/wave; live set ~100 fits.
__global__ __launch_bounds__(512) void rnn_recur(const float* __restrict__ w_hh,
                                                 float* __restrict__ out) {
  extern __shared__ __align__(16) char smem[];
  _Float16* sW = (_Float16*)smem;                       // [HDIM][LDK]  w_hh as [N][K], f16
  _Float16* sH = (_Float16*)(smem + HDIM * LDK * 2);    // [16][LDK]    h slice, f16
  const int tid  = threadIdx.x;
  const int lane = tid & 31;
  const int wave = tid >> 5;
  const int b0   = blockIdx.x * 16;

  for (int i = tid * 4; i < HDIM * HDIM; i += 512 * 4) {
    float4 v = *(const float4*)(w_hh + i);
    v4h h = {(_Float16)v.x, (_Float16)v.y, (_Float16)v.z, (_Float16)v.w};
    *(v4h*)(sW + (i >> 8) * LDK + (i & 255)) = h;
  }
  for (int i = tid * 4; i < 16 * LDK; i += 512 * 4) {   // h0 = 0
    v4h z = {};
    *(v4h*)(sH + i) = z;
  }
  __syncthreads();

  const int n0 = wave * 16;
  v16h Bf[8];                                            // step-invariant w_hh fragments
#pragma unroll
  for (int kb = 0; kb < 8; ++kb) Bf[kb] = ldsB(sW, n0, kb * 32, lane);

  const int n  = n0 + (lane & 15);
  const int rb = (lane >> 4) << 3;

  for (int t = 0; t < SEQ; ++t) {
    const size_t obase = (size_t)(t * BATCH + b0 + rb) * HDIM + n;

    // xw loads: no dependency on the WMMA chain -> issue early, overlap compute.
    v8f xw;
#pragma unroll
    for (int r = 0; r < 8; ++r) xw[r] = out[obase + (size_t)r * HDIM];

    v8f acc = {};                                        // zero C -> chain not blocked on loads
#pragma unroll
    for (int kb = 0; kb < 8; ++kb) {
      v16h a = ldsA(sH, 0, kb * 32, lane);
      acc = wmma_f16(a, Bf[kb], acc);
    }

    __syncthreads();                                     // all waves done reading h_{t-1}

#pragma unroll
    for (int r = 0; r < 8; ++r) {
      float hv = tanhf(acc[r] + xw[r]);
      out[obase + (size_t)r * HDIM] = hv;                // f32 result
      sH[(rb + r) * LDK + n] = (_Float16)hv;             // f16 h for next step
    }
    __syncthreads();                                     // h_t visible before next K-loop
  }
}

extern "C" void kernel_launch(void* const* d_in, const int* in_sizes, int n_in,
                              void* d_out, int out_size, void* d_ws, size_t ws_size,
                              hipStream_t stream) {
  const float* x    = (const float*)d_in[0];
  const float* w_ih = (const float*)d_in[1];
  const float* w_hh = (const float*)d_in[2];
  const float* b_ih = (const float*)d_in[3];
  const float* b_hh = (const float*)d_in[4];
  float* out = (float*)d_out;

  const int smem1 = (HDIM + 64) * LDK * 2;   // 168,960 B
  const int smem2 = (HDIM + 16) * LDK * 2;   // 143,616 B

  hipFuncSetAttribute(reinterpret_cast<const void*>(rnn_xproj),
                      hipFuncAttributeMaxDynamicSharedMemorySize, smem1);
  hipFuncSetAttribute(reinterpret_cast<const void*>(rnn_recur),
                      hipFuncAttributeMaxDynamicSharedMemorySize, smem2);

  rnn_xproj<<<(SEQ * BATCH) / 64, 256, smem1, stream>>>(x, w_ih, b_ih, b_hh, out);
  rnn_recur<<<BATCH / 16, 512, smem2, stream>>>(w_hh, out);
}
